// GraphTransformerModule_53326313947335
// MI455X (gfx1250) — matrix operand order repaired and checked by reading, not compiled, not run on hardware
//
#include <hip/hip_runtime.h>
#include <hip/hip_bf16.h>
#include <math.h>

// ---------------- CDNA5 WMMA types ----------------
typedef __bf16 bf16_t;
typedef __attribute__((ext_vector_type(16))) __bf16 v16bf;
typedef __attribute__((ext_vector_type(8)))  __bf16 v8bf;
typedef __attribute__((ext_vector_type(8)))  float  v8f;

#define DFEAT 128

__device__ __forceinline__ v16bf cat8(v8bf a, v8bf b) {
  return __builtin_shufflevector(a, b, 0,1,2,3,4,5,6,7,8,9,10,11,12,13,14,15);
}

// A-fragment (16x32 bf16) from a row-major LDS tile (stride in elements).
// Per ISA 7.12.2: lanes 0-15 hold K=0..7 (v0-3) and K=16..23 (v4-7);
// lanes 16-31 hold K=8..15 and K=24..31.
__device__ __forceinline__ v16bf load_a_frag(const bf16_t* tile, int stride, int lane, int kt) {
  int m = lane & 15, half = lane >> 4;
  const bf16_t* row = tile + m * stride + kt * 32 + half * 8;
  v8bf x0 = *(const v8bf*)(row);        // K = kt*32 + half*8 + [0..7]
  v8bf x1 = *(const v8bf*)(row + 16);   // K = kt*32 + 16 + half*8 + [0..7]
  return cat8(x0, x1);
}

// B-fragment from prepacked weights: 16 contiguous bf16 per lane.
__device__ __forceinline__ v16bf load_b_frag(const bf16_t* __restrict__ wp,
                                             int lane, int nt, int kt, int KT) {
  const bf16_t* p = wp + ((size_t)((nt * KT + kt) * 32 + lane) << 4);
  return cat8(*(const v8bf*)p, *(const v8bf*)(p + 8));
}

// One 16x16 output tile: D = A(16x(KT*32)) * W [+ bias]   (f32 accumulate)
// Bias presence is compile-time: no branches in the inner GEMM path.
template <int KT, bool HAS_BIAS>
__device__ __forceinline__ v8f gemm_tile(const v16bf* afr, const bf16_t* __restrict__ wp,
                                         const float* __restrict__ bias, int lane, int nt) {
  float b = 0.0f;
  if (HAS_BIAS) b = bias[nt * 16 + (lane & 15)];
  v8f acc = {b, b, b, b, b, b, b, b};
#pragma unroll
  for (int kt = 0; kt < KT; ++kt) {
    v16bf bf = load_b_frag(wp, lane, nt, kt, KT);
    acc = __builtin_amdgcn_wmma_f32_16x16x32_bf16(false, afr[kt], false, bf,
                                                  (short)0, acc, false, false);
  }
  return acc;
}

// LayerNorm one 128-float row -> bf16 into LDS (two passes, f32 stats).
__device__ __forceinline__ void ln_row_to_lds(const float* __restrict__ x,
                                              const float* __restrict__ w,
                                              const float* __restrict__ b,
                                              bf16_t* out) {
  float s = 0.f, s2 = 0.f;
#pragma unroll
  for (int i = 0; i < DFEAT; i += 4) {
    float4 t = *(const float4*)(x + i);
    s  += t.x + t.y + t.z + t.w;
    s2 += t.x * t.x + t.y * t.y + t.z * t.z + t.w * t.w;
  }
  float mn  = s * (1.f / 128.f);
  float inv = rsqrtf(s2 * (1.f / 128.f) - mn * mn + 1e-5f);
#pragma unroll
  for (int i = 0; i < DFEAT; i += 4) {
    float4 t  = *(const float4*)(x + i);
    float4 ww = *(const float4*)(w + i);
    float4 bb = *(const float4*)(b + i);
    out[i + 0] = (bf16_t)((t.x - mn) * inv * ww.x + bb.x);
    out[i + 1] = (bf16_t)((t.y - mn) * inv * ww.y + bb.y);
    out[i + 2] = (bf16_t)((t.z - mn) * inv * ww.z + bb.z);
    out[i + 3] = (bf16_t)((t.w - mn) * inv * ww.w + bb.w);
  }
}

// Shared tail: x2 = T@Wo + bo + resid ; LN2 (in-register) ; FFN 128->256 SiLU ->128 ; out = x2 + ffn
// T: per-wave 16-row LDS tile, stride 256 bf16. Caller must __syncthreads() before.
__device__ __forceinline__ void tail_block(bf16_t* T, int lane,
                                           const bf16_t* __restrict__ wWo, const float* __restrict__ bo,
                                           const float* __restrict__ resid,
                                           const float* __restrict__ l2w, const float* __restrict__ l2b,
                                           const bf16_t* __restrict__ w1, const bf16_t* __restrict__ w2,
                                           float* __restrict__ out, long rbase, int nvalid) {
  const int S = 256;
  int half = lane >> 4, cl = lane & 15;
  v16bf afr[4];
#pragma unroll
  for (int kt = 0; kt < 4; ++kt) afr[kt] = load_a_frag(T, S, lane, kt);

  v8f e2[8];
#pragma unroll
  for (int nt = 0; nt < 8; ++nt) {
    v8f a = gemm_tile<4, true>(afr, wWo, bo, lane, nt);
    int col = nt * 16 + cl;
#pragma unroll
    for (int r = 0; r < 8; ++r) {
      int m = r + half * 8;
      if (m < nvalid) a[r] += resid[(rbase + m) * DFEAT + col];
    }
    e2[nt] = a;
  }
  // LN2 on D-layout registers: per-row mean/var via 16-lane shfl_xor reduce
  float mean[8], inv[8];
#pragma unroll
  for (int r = 0; r < 8; ++r) {
    float s = 0.f, s2 = 0.f;
#pragma unroll
    for (int nt = 0; nt < 8; ++nt) { float xv = e2[nt][r]; s += xv; s2 += xv * xv; }
#pragma unroll
    for (int msk = 1; msk < 16; msk <<= 1) {
      s  += __shfl_xor(s,  msk, 32);
      s2 += __shfl_xor(s2, msk, 32);
    }
    float mn = s * (1.f / 128.f);
    mean[r] = mn;
    inv[r]  = rsqrtf(s2 * (1.f / 128.f) - mn * mn + 1e-5f);
  }
#pragma unroll
  for (int nt = 0; nt < 8; ++nt) {
    int col = nt * 16 + cl;
    float wc = l2w[col], bc = l2b[col];
#pragma unroll
    for (int r = 0; r < 8; ++r)
      T[(r + half * 8) * S + col] = (bf16_t)((e2[nt][r] - mean[r]) * inv[r] * wc + bc);
  }
  __syncthreads();
  // FFN up (128 -> 256) + SiLU, hidden kept in LDS (bf16)
#pragma unroll
  for (int kt = 0; kt < 4; ++kt) afr[kt] = load_a_frag(T, S, lane, kt);
#pragma unroll
  for (int nt = 0; nt < 16; ++nt) {
    v8f h = gemm_tile<4, false>(afr, w1, nullptr, lane, nt);
    int col = nt * 16 + cl;
#pragma unroll
    for (int r = 0; r < 8; ++r) {
      float xv = h[r];
      T[(r + half * 8) * S + col] = (bf16_t)(xv / (1.f + __expf(-xv)));
    }
  }
  __syncthreads();
  // FFN down (256 -> 128), add residual x2, store
  v16bf af2[8];
#pragma unroll
  for (int kt = 0; kt < 8; ++kt) af2[kt] = load_a_frag(T, S, lane, kt);
#pragma unroll
  for (int nt = 0; nt < 8; ++nt) {
    v8f a = gemm_tile<8, false>(af2, w2, nullptr, lane, nt);
    int col = nt * 16 + cl;
#pragma unroll
    for (int r = 0; r < 8; ++r) {
      int m = r + half * 8;
      if (m < nvalid) out[(rbase + m) * DFEAT + col] = a[r] + e2[nt][r];
    }
  }
}

// ---------------- Kernel 0: zero f32 region ----------------
__global__ void k_zero(float* p, long n) {
  long i = (long)blockIdx.x * blockDim.x + threadIdx.x;
  long st = (long)gridDim.x * blockDim.x;
  for (; i < n; i += st) p[i] = 0.f;
}

// ---------------- Kernel 1: prepack f32 weights -> bf16 B-fragments ----------------
__global__ void k_prepack(const float* __restrict__ W, bf16_t* __restrict__ out,
                          int KT, int ncols, int total) {
  int idx = blockIdx.x * blockDim.x + threadIdx.x;
  if (idx >= total) return;
  int e    = idx & 15;
  int lane = (idx >> 4) & 31;
  int t    = idx >> 9;
  int kt   = t % KT;
  int nt   = t / KT;
  int K    = kt * 32 + (e >> 3) * 16 + (lane >> 4) * 8 + (e & 7);      // B-layout: lanes 0-15 K 0-7,16-23; lanes 16-31 K 8-15,24-31
  int col  = nt * 16 + (lane & 15);
  out[idx] = (bf16_t)W[(size_t)K * ncols + col];
}

// ---------------- Kernel 2: node LN1 + Q/K/V projections ----------------
__global__ __launch_bounds__(256) void k_node_qkv(
    const float* __restrict__ x, const float* l1w, const float* l1b,
    const bf16_t* __restrict__ wq, const float* __restrict__ bq,
    const bf16_t* __restrict__ wk, const float* __restrict__ bk,
    const bf16_t* __restrict__ wv, const float* __restrict__ bv,
    float* __restrict__ Qh, float* __restrict__ Kh, float* __restrict__ Vh, int nrows) {
  __shared__ bf16_t L[128 * DFEAT];
  int tid = threadIdx.x, base = blockIdx.x * 128;
  if (tid < 128) {
    int row = base + tid;
    bf16_t* lr = L + tid * DFEAT;
    if (row < nrows) {
      ln_row_to_lds(x + (size_t)row * DFEAT, l1w, l1b, lr);
    } else {
#pragma unroll
      for (int c = 0; c < DFEAT; ++c) lr[c] = (bf16_t)0.f;
    }
  }
  __syncthreads();
  int w = tid >> 5, lane = tid & 31, half = lane >> 4, cl = lane & 15;
  const bf16_t* T = L + (w * 16) * DFEAT;
  v16bf afr[4];
#pragma unroll
  for (int kt = 0; kt < 4; ++kt) afr[kt] = load_a_frag(T, DFEAT, lane, kt);
#pragma unroll
  for (int nt = 0; nt < 8; ++nt) {
    v8f aq = gemm_tile<4, true>(afr, wq, bq, lane, nt);
    v8f ak = gemm_tile<4, true>(afr, wk, bk, lane, nt);
    v8f av = gemm_tile<4, true>(afr, wv, bv, lane, nt);
    int col = nt * 16 + cl;
#pragma unroll
    for (int r = 0; r < 8; ++r) {
      int row = base + w * 16 + r + half * 8;
      if (row < nrows) {
        size_t o = (size_t)row * DFEAT + col;
        Qh[o] = aq[r]; Kh[o] = ak[r]; Vh[o] = av[r];
      }
    }
  }
}

// ---------------- Kernel 3: fused edge pipeline ----------------
// LN1(e) -> proj_e = @We+be -> score/e_out -> per-head w -> atomic scatter wV,z
// -> e2 = e_out@Wo_e + bo_e + e_in -> LN2 -> FFN -> e_out final
__global__ __launch_bounds__(256) void k_edge(
    const float* __restrict__ ef, const int* __restrict__ src, const int* __restrict__ dst,
    const float* l1w, const float* l1b, const float* l2w, const float* l2b,
    const bf16_t* __restrict__ wWe, const float* __restrict__ be,
    const bf16_t* __restrict__ wWo, const float* __restrict__ bo,
    const bf16_t* __restrict__ w1, const bf16_t* __restrict__ w2,
    const float* __restrict__ Kh, const float* __restrict__ Qh, const float* __restrict__ Vh,
    float* __restrict__ wVacc, float* __restrict__ zacc, float* __restrict__ eout) {
  __shared__ bf16_t L[128 * 256];
  const int S = 256;
  int tid = threadIdx.x, base = blockIdx.x * 128;
  __builtin_prefetch(wWe, 0, 0);
  __builtin_prefetch(w1, 0, 0);
  if (tid < 128) ln_row_to_lds(ef + (size_t)(base + tid) * DFEAT, l1w, l1b, L + tid * S);
  __syncthreads();

  int w = tid >> 5, lane = tid & 31, half = lane >> 4, cl = lane & 15;
  bf16_t* T = L + (w * 16) * S;
  long rbase = base + w * 16;

  int sR[8], dR[8];
#pragma unroll
  for (int r = 0; r < 8; ++r) {
    int m = r + half * 8;
    sR[r] = src[rbase + m];
    dR[r] = dst[rbase + m];
  }
  v16bf afr[4];
#pragma unroll
  for (int kt = 0; kt < 4; ++kt) afr[kt] = load_a_frag(T, S, lane, kt);

  // head tile nt == head index (DH == 16)
#pragma unroll
  for (int nt = 0; nt < 8; ++nt) {
    v8f p = gemm_tile<4, true>(afr, wWe, be, lane, nt);   // proj_e tile
    int col = nt * 16 + cl;
    float eo[8], wg[8];
#pragma unroll
    for (int r = 0; r < 8; ++r) {
      float kv = Kh[(size_t)sR[r] * DFEAT + col];   // L2-resident gathers
      float qv = Qh[(size_t)dR[r] * DFEAT + col];
      float sc = fminf(fmaxf(kv * qv * 0.25f, -5.f), 5.f);
      eo[r] = sc * p[r];
    }
#pragma unroll
    for (int r = 0; r < 8; ++r) {
      float s = eo[r];
      s += __shfl_xor(s, 1, 32);
      s += __shfl_xor(s, 2, 32);
      s += __shfl_xor(s, 4, 32);
      s += __shfl_xor(s, 8, 32);
      wg[r] = __expf(fminf(fmaxf(s, -5.f), 5.f));
    }
#pragma unroll
    for (int r = 0; r < 8; ++r) {
      float vv = Vh[(size_t)sR[r] * DFEAT + col];
      atomicAdd(wVacc + (size_t)dR[r] * DFEAT + col, wg[r] * vv);
    }
    if (cl == 0) {
#pragma unroll
      for (int r = 0; r < 8; ++r) atomicAdd(zacc + (size_t)dR[r] * 8 + nt, wg[r]);
    }
#pragma unroll
    for (int r = 0; r < 8; ++r)
      T[(r + half * 8) * S + col] = (bf16_t)eo[r];   // e_attn for Wo_e GEMM
  }
  __syncthreads();
  tail_block(T, lane, wWo, bo, ef, l2w, l2b, w1, w2, eout, rbase, 16);
}

// ---------------- Kernel 4: node finalize ----------------
// h_attn = wV/(z+eps) -> @Wo_n+bo_n + h_in -> LN2 -> FFN -> h_out
__global__ __launch_bounds__(256) void k_node_out(
    const float* __restrict__ xin, const float* __restrict__ wVacc, const float* __restrict__ zacc,
    const float* __restrict__ l2w, const float* __restrict__ l2b,
    const bf16_t* __restrict__ wWo, const float* __restrict__ bo,
    const bf16_t* __restrict__ w1, const bf16_t* __restrict__ w2,
    float* __restrict__ hout, int nrows) {
  __shared__ bf16_t L[128 * 256];
  const int S = 256;
  int tid = threadIdx.x, base = blockIdx.x * 128;
  if (tid < 128) {
    int row = base + tid;
    bf16_t* lr = L + tid * S;
    if (row < nrows) {
#pragma unroll
      for (int c = 0; c < DFEAT; ++c) {
        float z = zacc[(size_t)row * 8 + (c >> 4)];
        float v = wVacc[(size_t)row * DFEAT + c];
        lr[c] = (bf16_t)(v / (z + 1e-8f));
      }
    } else {
#pragma unroll
      for (int c = 0; c < DFEAT; ++c) lr[c] = (bf16_t)0.f;
    }
  }
  __syncthreads();
  int w = tid >> 5, lane = tid & 31;
  bf16_t* T = L + (w * 16) * S;
  long rbase = base + w * 16;
  int nvalid = nrows - (int)rbase;
  if (nvalid > 16) nvalid = 16;
  if (nvalid < 0) nvalid = 0;
  tail_block(T, lane, wWo, bo, xin, l2w, l2b, w1, w2, hout, rbase, nvalid);
}

// ---------------- launch ----------------
extern "C" void kernel_launch(void* const* d_in, const int* in_sizes, int n_in,
                              void* d_out, int out_size, void* d_ws, size_t ws_size,
                              hipStream_t stream) {
  const float* node_feats = (const float*)d_in[0];
  const float* edge_feats = (const float*)d_in[1];
  const int*   e_src      = (const int*)d_in[2];
  const int*   e_dst      = (const int*)d_in[3];
  const float* ln1n_w = (const float*)d_in[4];  const float* ln1n_b = (const float*)d_in[5];
  const float* ln1e_w = (const float*)d_in[6];  const float* ln1e_b = (const float*)d_in[7];
  const float* ln2n_w = (const float*)d_in[8];  const float* ln2n_b = (const float*)d_in[9];
  const float* ln2e_w = (const float*)d_in[10]; const float* ln2e_b = (const float*)d_in[11];
  const float* Wq = (const float*)d_in[12]; const float* bq = (const float*)d_in[13];
  const float* Wk = (const float*)d_in[14]; const float* bk = (const float*)d_in[15];
  const float* Wv = (const float*)d_in[16]; const float* bv = (const float*)d_in[17];
  const float* We = (const float*)d_in[18]; const float* be = (const float*)d_in[19];
  const float* Wo_n = (const float*)d_in[20]; const float* bo_n = (const float*)d_in[21];
  const float* Wo_e = (const float*)d_in[22]; const float* bo_e = (const float*)d_in[23];
  const float* mlp_n_w1 = (const float*)d_in[24];
  const float* mlp_n_w2 = (const float*)d_in[25];
  const float* mlp_e_w1 = (const float*)d_in[26];
  const float* mlp_e_w2 = (const float*)d_in[27];

  int nN = in_sizes[0] / DFEAT;
  int nE = in_sizes[2];

  char* ws = (char*)d_ws;
  size_t off = 0;
  auto take = [&](size_t bytes) -> char* {
    char* p = ws + off;
    off = (off + bytes + 255) & ~(size_t)255;
    return p;
  };
  bf16_t* pWq  = (bf16_t*)take(16384 * 2);
  bf16_t* pWk  = (bf16_t*)take(16384 * 2);
  bf16_t* pWv  = (bf16_t*)take(16384 * 2);
  bf16_t* pWe  = (bf16_t*)take(16384 * 2);
  bf16_t* pWon = (bf16_t*)take(16384 * 2);
  bf16_t* pWoe = (bf16_t*)take(16384 * 2);
  bf16_t* pW1n = (bf16_t*)take(32768 * 2);
  bf16_t* pW2n = (bf16_t*)take(32768 * 2);
  bf16_t* pW1e = (bf16_t*)take(32768 * 2);
  bf16_t* pW2e = (bf16_t*)take(32768 * 2);
  float* Qh = (float*)take((size_t)nN * DFEAT * 4);
  float* Kh = (float*)take((size_t)nN * DFEAT * 4);
  float* Vh = (float*)take((size_t)nN * DFEAT * 4);
  float* wV = (float*)take((size_t)nN * DFEAT * 4);
  float* zS = (float*)take((size_t)nN * 8 * 4);

  // 1) zero scatter accumulators (required every launch; harness does not re-poison)
  k_zero<<<2048, 256, 0, stream>>>(wV, (long)nN * DFEAT);
  k_zero<<<256, 256, 0, stream>>>(zS, (long)nN * 8);

  // 2) prepack weights into bf16 WMMA B-fragment layout
  k_prepack<<<64, 256, 0, stream>>>(Wq, pWq, 4, 128, 16384);
  k_prepack<<<64, 256, 0, stream>>>(Wk, pWk, 4, 128, 16384);
  k_prepack<<<64, 256, 0, stream>>>(Wv, pWv, 4, 128, 16384);
  k_prepack<<<64, 256, 0, stream>>>(We, pWe, 4, 128, 16384);
  k_prepack<<<64, 256, 0, stream>>>(Wo_n, pWon, 4, 128, 16384);
  k_prepack<<<64, 256, 0, stream>>>(Wo_e, pWoe, 4, 128, 16384);
  k_prepack<<<128, 256, 0, stream>>>(mlp_n_w1, pW1n, 4, 256, 32768);
  k_prepack<<<128, 256, 0, stream>>>(mlp_n_w2, pW2n, 8, 128, 32768);
  k_prepack<<<128, 256, 0, stream>>>(mlp_e_w1, pW1e, 4, 256, 32768);
  k_prepack<<<128, 256, 0, stream>>>(mlp_e_w2, pW2e, 8, 128, 32768);

  int gN = (nN + 127) / 128;
  int gE = (nE + 127) / 128;

  // 3) node LN1 + QKV (WMMA)
  k_node_qkv<<<gN, 256, 0, stream>>>(node_feats, ln1n_w, ln1n_b,
                                     pWq, bq, pWk, bk, pWv, bv, Qh, Kh, Vh, nN);

  // 4) fused edge pipeline (single pass over 410 MB edge tensor)
  float* e_out = (float*)d_out + (size_t)nN * DFEAT;
  k_edge<<<gE, 256, 0, stream>>>(edge_feats, e_src, e_dst,
                                 ln1e_w, ln1e_b, ln2e_w, ln2e_b,
                                 pWe, be, pWoe, bo_e, pW1e, pW2e,
                                 Kh, Qh, Vh, wV, zS, e_out);

  // 5) node finalize (attention normalize + Wo_n + LN2 + FFN)
  k_node_out<<<gN, 256, 0, stream>>>(node_feats, wV, zS, ln2n_w, ln2n_b,
                                     pWon, bo_n, pW1n, pW2n, (float*)d_out, nN);
}
